// ChunkingLayer_89472758710928
// MI455X (gfx1250) — matrix operand order repaired and checked by reading, not compiled, not run on hardware
//
#include <hip/hip_runtime.h>
#include <hip/hip_bf16.h>

// Problem constants (match reference)
#define BB 8
#define SS 4096
#define DD 1024

#define ROWS_WG 128                     // rows per workgroup (8 waves x 16)
#define STAGE_ROWS (ROWS_WG + 1)        // +1 row feeds the roll-by-1 K path
#define STAGE_BYTES (STAGE_ROWS * DD * (int)sizeof(_Float16))  // 264192 B

typedef __attribute__((ext_vector_type(16))) _Float16 v16h;
typedef __attribute__((ext_vector_type(8)))  _Float16 v8h;
typedef __attribute__((ext_vector_type(8)))  float    v8f;

// ---------------------------------------------------------------------------
// Kernel 0: convert Wq/Wk (f32 row-major [e][d]) to f16 in workspace.
// 4 MB total -> stays resident in the 192MB L2 for the whole GEMM sweep.
// ---------------------------------------------------------------------------
__global__ __launch_bounds__(256) void convert_w_kernel(
    const float* __restrict__ wq, const float* __restrict__ wk,
    _Float16* __restrict__ oq, _Float16* __restrict__ ok) {
  int i = blockIdx.x * 256 + threadIdx.x;
  if (i < DD * DD) {
    oq[i] = (_Float16)wq[i];
    ok[i] = (_Float16)wk[i];
  }
}

// ---------------------------------------------------------------------------
// Kernel 1: fused routing GEMM. A workgroup (8 waves) owns 128 sequence rows
// of one batch; 129 x-rows are staged once into dynamic LDS as f16 (264 KB of
// the 320 KB WGP LDS). Wave w computes the 16-row M-tile [w*16, w*16+16) with
// v_wmma_f32_16x16x32_f16 pairs (Q and shifted-K share the staged A data one
// LDS row apart). All 8 waves sweep (e0, d0) in lockstep (barrier per N-tile)
// so their identical Wq/Wk loads hit the WGP$ — B traffic from L2 drops 8x
// versus per-wave workgroups (~1 GB total, ~128 FLOP/byte from L2).
// Q/K are never materialized: only per-row dot(q,k), |q|^2, |k|^2 survive.
// ---------------------------------------------------------------------------
__global__ __launch_bounds__(256) void routing_kernel(
    const float* __restrict__ x,
    const _Float16* __restrict__ wq, const _Float16* __restrict__ wk,
    const float* __restrict__ bq, const float* __restrict__ bk,
    float* __restrict__ p_out, float* __restrict__ b_out) {
  extern __shared__ __align__(16) _Float16 xs[];   // STAGE_ROWS * DD halfs

  const int blk  = blockIdx.x;                 // 0 .. B*S/128 - 1
  const int bidx = blk / (SS / ROWS_WG);
  const int s0   = (blk % (SS / ROWS_WG)) * ROWS_WG;
  const int tid  = threadIdx.x;
  const int wave = tid >> 5;                   // 0..7
  const int lane = tid & 31;                   // wave32
  const int half = lane >> 4;                  // lane group 0/1
  const int n    = lane & 15;                  // A-row (M) and B/C column (N)
  const int m0   = wave * 16;                  // this wave's M-tile base row

  const float* xb = x + (size_t)bidx * SS * DD;

  // Stage rows s0-1 .. s0+127 (roll wraps inside the batch) as f16, once.
  for (int idx = tid; idx < STAGE_ROWS * DD; idx += 256) {
    int r = idx >> 10;
    int d = idx & (DD - 1);
    int s = s0 - 1 + r;
    if (s < 0) s += SS;                        // jnp.roll wrap at s==0
    xs[idx] = (_Float16)xb[(size_t)s * DD + d];
  }
  __syncthreads();

  float dot[8], qq[8], kk[8];
#pragma unroll
  for (int r = 0; r < 8; ++r) { dot[r] = 0.f; qq[r] = 0.f; kk[r] = 0.f; }

  // A layout (contiguous, matches B): lane half h holds K = 16h..16h+15
  const int a_off = half << 4;   // 0 or 16
  // B layout (contiguous, matches A): lanes 0-15 -> K 0..15, lanes 16-31 -> 16..31
  const int b_off = half << 4;

  const _Float16* aq_base = &xs[(m0 + n + 1) * DD];  // Q uses x[s]
  const _Float16* ak_base = &xs[(m0 + n) * DD];      // K uses x[s-1]

  for (int e0 = 0; e0 < DD; e0 += 16) {
    v8f accq = {};
    v8f acck = {};
    const _Float16* wqrow = wq + (size_t)(e0 + n) * DD;
    const _Float16* wkrow = wk + (size_t)(e0 + n) * DD;
    if (e0 + 16 < DD) {                        // hint next N-tile's W rows
      __builtin_prefetch(wqrow + (size_t)16 * DD, 0, 1);
      __builtin_prefetch(wkrow + (size_t)16 * DD, 0, 1);
    }

#pragma unroll 4
    for (int d0 = 0; d0 < DD; d0 += 32) {
      const int ab = d0 + a_off;
      const _Float16* aq = aq_base + ab;
      const _Float16* ak = ak_base + ab;
      v8h aql = *(const v8h*)aq;
      v8h aqh = *(const v8h*)(aq + 8);
      v8h akl = *(const v8h*)ak;
      v8h akh = *(const v8h*)(ak + 8);
      v16h Aq = __builtin_shufflevector(aql, aqh, 0,1,2,3,4,5,6,7,8,9,10,11,12,13,14,15);
      v16h Ak = __builtin_shufflevector(akl, akh, 0,1,2,3,4,5,6,7,8,9,10,11,12,13,14,15);

      v16h Bq = *(const v16h*)(wqrow + d0 + b_off);   // 32B contiguous, L1/L2-hot
      v16h Bk = *(const v16h*)(wkrow + d0 + b_off);

      accq = __builtin_amdgcn_wmma_f32_16x16x32_f16(false, Aq, false, Bq,
                                                    (short)0, accq, false, false);
      acck = __builtin_amdgcn_wmma_f32_16x16x32_f16(false, Ak, false, Bk,
                                                    (short)0, acck, false, false);
    }

    const float bqv = bq[e0 + n];
    const float bkv = bk[e0 + n];
#pragma unroll
    for (int r = 0; r < 8; ++r) {       // C/D layout: VGPR r = row r+8*half, col n
      float qv = accq[r] + bqv;
      float kv = acck[r] + bkv;
      dot[r] = fmaf(qv, kv, dot[r]);
      qq[r]  = fmaf(qv, qv, qq[r]);
      kk[r]  = fmaf(kv, kv, kk[r]);
    }
    __syncthreads();                    // keep the 8 waves' B streams in step
  }

  // Reduce over the 16 columns held by each 16-lane group.
#pragma unroll
  for (int m = 1; m <= 8; m <<= 1) {
#pragma unroll
    for (int r = 0; r < 8; ++r) {
      dot[r] += __shfl_xor(dot[r], m, 32);
      qq[r]  += __shfl_xor(qq[r],  m, 32);
      kk[r]  += __shfl_xor(kk[r],  m, 32);
    }
  }

  if (n == 0) {                          // lanes 0 and 16 each emit 8 rows
    const int rowbase = s0 + m0 + half * 8;
#pragma unroll
    for (int r = 0; r < 8; ++r) {
      int s = rowbase + r;
      float dq = fmaxf(sqrtf(qq[r]), 1e-8f);   // torch cosine_similarity eps
      float dk = fmaxf(sqrtf(kk[r]), 1e-8f);
      float cs = dot[r] / (dq * dk);
      float p  = 0.5f * (1.0f - cs);
      if (s == 0) p = 1.0f;                    // p[:,0] = 1
      p_out[bidx * SS + s] = p;
      b_out[bidx * SS + s] = (p >= 0.5f) ? 1.0f : 0.0f;
    }
  }
}

// ---------------------------------------------------------------------------
// Kernel 2: per-batch stable compaction plan. Exclusive prefix-sum of flags,
// emit inverse map (dest slot -> source s) and counts. One block per batch.
// ---------------------------------------------------------------------------
__global__ __launch_bounds__(256) void scan_kernel(
    const float* __restrict__ flags, int* __restrict__ inv,
    int* __restrict__ counts) {
  __shared__ int sums[256];
  __shared__ int offs[256];
  const int bidx = blockIdx.x;
  const int t = threadIdx.x;
  const float* f = flags + bidx * SS;
  const int base = t * 16;                    // 256 * 16 = 4096 = S

  int c = 0;
#pragma unroll
  for (int j = 0; j < 16; ++j) c += (f[base + j] >= 0.5f) ? 1 : 0;
  sums[t] = c;
  __syncthreads();
  if (t == 0) {                               // tiny serial scan (256 adds)
    int acc = 0;
    for (int i = 0; i < 256; ++i) { offs[i] = acc; acc += sums[i]; }
    counts[bidx] = acc;
  }
  __syncthreads();
  int run = offs[t];
#pragma unroll
  for (int j = 0; j < 16; ++j) {
    if (f[base + j] >= 0.5f) {
      inv[bidx * SS + run] = base + j;        // stable: ascending s
      ++run;
    }
  }
}

// ---------------------------------------------------------------------------
// Kernel 3: write every destination row exactly once (single 134MB stream):
// copy x[source] if slot < count, else zeros. One block per destination row,
// 256 threads * float4 = 1024 floats.
// ---------------------------------------------------------------------------
__global__ __launch_bounds__(256) void gather_kernel(
    const float* __restrict__ x, const int* __restrict__ inv,
    const int* __restrict__ counts, float* __restrict__ out) {
  const int row  = blockIdx.x;                // 0 .. B*S-1
  const int bidx = row / SS;
  const int j    = row - bidx * SS;
  const int t    = threadIdx.x;

  float4 v = make_float4(0.f, 0.f, 0.f, 0.f);
  if (j < counts[bidx]) {
    int s = inv[bidx * SS + j];
    v = ((const float4*)(x + ((size_t)bidx * SS + s) * DD))[t];
  }
  ((float4*)(out + (size_t)row * DD))[t] = v;
}

// ---------------------------------------------------------------------------
extern "C" void kernel_launch(void* const* d_in, const int* in_sizes, int n_in,
                              void* d_out, int out_size, void* d_ws, size_t ws_size,
                              hipStream_t stream) {
  const float* x  = (const float*)d_in[0];
  const float* Wq = (const float*)d_in[1];
  const float* bq = (const float*)d_in[2];
  const float* Wk = (const float*)d_in[3];
  const float* bk = (const float*)d_in[4];

  // Output: selected_padded [B,S,D] ++ p [B,S] ++ b [B,S], all f32.
  float* out_sel = (float*)d_out;
  float* out_p   = out_sel + (size_t)BB * SS * DD;
  float* out_b   = out_p + (size_t)BB * SS;

  // Workspace: f16 Wq (2MB) | f16 Wk (2MB) | inverse map (128KB) | counts
  _Float16* wsq = (_Float16*)d_ws;
  _Float16* wsk = wsq + (size_t)DD * DD;
  int* inv      = (int*)(wsk + (size_t)DD * DD);
  int* counts   = inv + BB * SS;

  // Opt in to >64KB dynamic LDS (idempotent, deterministic, capture-safe).
  (void)hipFuncSetAttribute((const void*)routing_kernel,
                            hipFuncAttributeMaxDynamicSharedMemorySize,
                            STAGE_BYTES);

  convert_w_kernel<<<(DD * DD + 255) / 256, 256, 0, stream>>>(Wq, Wk, wsq, wsk);
  routing_kernel<<<BB * SS / ROWS_WG, 256, STAGE_BYTES, stream>>>(
      x, wsq, wsk, bq, bk, out_p, out_b);
  scan_kernel<<<BB, 256, 0, stream>>>(out_b, inv, counts);
  gather_kernel<<<BB * SS, 256, 0, stream>>>(x, inv, counts, out_sel);
}
